// GCN_60215441490202
// MI455X (gfx1250) — matrix operand order repaired and checked by reading, not compiled, not run on hardware
//
#include <hip/hip_runtime.h>
#include <hip/hip_bf16.h>

#define NN 50000
#define EE 600000
#define GG 512
#define MM 2048
#define DD 128
#define FFN 256
#define LL 3
#define BN_EPS 1e-5f

typedef __attribute__((ext_vector_type(16))) __bf16 v16bf;
typedef __attribute__((ext_vector_type(8)))  float  v8f;

union FragU { uint4 u[2]; v16bf v; };

#define CDIV(a,b) (((a)+(b)-1)/(b))

// ---------------- utility kernels ----------------

__global__ void k_zero_f32(float* p, size_t n) {
    size_t i = (size_t)blockIdx.x * blockDim.x + threadIdx.x;
    if (i < n) p[i] = 0.0f;
}

__global__ void k_deg_accum(const int* __restrict__ src, const int* __restrict__ dst,
                            float* deg_out, float* deg_in, int E) {
    int e = blockIdx.x * blockDim.x + threadIdx.x;
    if (e < E) {
        atomicAdd(&deg_out[src[e]], 1.0f);
        atomicAdd(&deg_in[dst[e]], 1.0f);
    }
}

__global__ void k_rsqrt_clip(float* p, int n) {
    int i = blockIdx.x * blockDim.x + threadIdx.x;
    if (i < n) p[i] = rsqrtf(fmaxf(p[i], 1.0f));
}

// W[K][Nc] (fp32 row-major) -> WT[Nc][K] (bf16, column-of-W contiguous)
__global__ void k_transpose_bf16(const float* __restrict__ W, __bf16* __restrict__ WT,
                                 int K, int Nc) {
    int idx = blockIdx.x * blockDim.x + threadIdx.x;
    if (idx >= K * Nc) return;
    int k = idx / Nc, n = idx % Nc;
    WT[(size_t)n * K + k] = (__bf16)W[idx];
}

// h -> bf16 copies: plain and row-scaled by norm_src
__global__ void k_prep_h(const float* __restrict__ hin, const float* __restrict__ norm_src,
                         __bf16* __restrict__ hb, __bf16* __restrict__ hsb, int total) {
    int idx = blockIdx.x * blockDim.x + threadIdx.x;
    if (idx >= total) return;
    int n = idx >> 7; // /128
    float v = hin[idx];
    hb[idx] = (__bf16)v;
    hsb[idx] = (__bf16)(v * norm_src[n]);
}

// ---------------- WMMA GEMM with register-resident B ----------------
// C[M,N] = A_bf16[M,K] @ BT_bf16[N,K] (+bias, optional relu, optional bf16 mirror).
// One wave owns one 16-wide column tile; loads ALL B fragments for its column
// into registers once, then sweeps MT consecutive 16-row tiles reusing them.
// Per-wave-uniform control flow only => EXEC stays all-ones for WMMA.
template <int KSTEPS, int MT, bool RELU, bool HASCB>
__global__ void k_wmma_gemm_t(const __bf16* __restrict__ A, const __bf16* __restrict__ BT,
                              const float* __restrict__ bias, float* __restrict__ C,
                              __bf16* __restrict__ Cb, int Mtiles, int Ncols) {
    constexpr int K = KSTEPS * 32;
    int ntn = Ncols >> 4;
    int wave = blockIdx.x * (blockDim.x >> 5) + (threadIdx.x >> 5);
    int mgroups = (Mtiles + MT - 1) / MT;
    int total = mgroups * ntn;
    if (wave >= total) return;
    int tg = wave / ntn, tn = wave % ntn;
    int lane = threadIdx.x & 31;
    int half = lane >> 4, r = lane & 15;

    // B fragments for this column tile, register-resident across the M sweep.
    const __bf16* Bcol = BT + (size_t)(tn * 16 + r) * K;
    FragU b[KSTEPS];
#pragma unroll
    for (int ks = 0; ks < KSTEPS; ++ks) {
        const uint4* bp = (const uint4*)(Bcol + ks * 32 + half * 16);
        b[ks].u[0] = bp[0];
        b[ks].u[1] = bp[1];
    }

    int n = tn * 16 + r;
    float bv = bias ? bias[n] : 0.0f;

    int m_end = Mtiles < (tg + 1) * MT ? Mtiles : (tg + 1) * MT;
    for (int tm = tg * MT; tm < m_end; ++tm) {
        const __bf16* Arow = A + (size_t)(tm * 16 + r) * K;
        if (tm + 1 < m_end)
            __builtin_prefetch(Arow + K * 16, 0, 1);   // next M-tile, same lane-row

        // Preload the whole A tile first so the loads clause together and the
        // waits stagger across the WMMA chain instead of lock-stepping.
        FragU a[KSTEPS];
#pragma unroll
        for (int ks = 0; ks < KSTEPS; ++ks) {
            a[ks].u[0] = *(const uint4*)(Arow + ks * 32 + half * 16);
            a[ks].u[1] = *(const uint4*)(Arow + ks * 32 + half * 16 + 8);
        }

        v8f acc = {};
#pragma unroll
        for (int ks = 0; ks < KSTEPS; ++ks) {
            acc = __builtin_amdgcn_wmma_f32_16x16x32_bf16(false, a[ks].v, false, b[ks].v,
                                                          (short)0, acc, false, false);
        }

#pragma unroll
        for (int v = 0; v < 8; ++v) {
            int m = tm * 16 + half * 8 + v;
            float x = acc[v] + bv;
            if (RELU) x = fmaxf(x, 0.0f);
            C[(size_t)m * Ncols + n] = x;
            if (HASCB) Cb[(size_t)m * Ncols + n] = (__bf16)x;
        }
    }
}

// ---------------- edge scatter: agg[dst] += hw[src] ----------------
__global__ void k_edge_scatter(const int* __restrict__ src, const int* __restrict__ dst,
                               const float* __restrict__ hw, float* __restrict__ agg, int E) {
    int t = blockIdx.x * blockDim.x + threadIdx.x;
    int e = t >> 5;
    if (e >= E) return;
    int j = (t & 31) * 4;
    int s = src[e], d = dst[e];
    float4 v = *(const float4*)(hw + (size_t)s * DD + j);
    float* ap = agg + (size_t)d * DD + j;
    atomicAdd(ap + 0, v.x);
    atomicAdd(ap + 1, v.y);
    atomicAdd(ap + 2, v.z);
    atomicAdd(ap + 3, v.w);
}

// ---------------- y = relu(agg*norm_dst + bg) + res ; BN partial sums ----------------
__global__ void k_combine_bnstats(const float* __restrict__ agg, const float* __restrict__ res,
                                  const float* __restrict__ norm_dst, const float* __restrict__ bg,
                                  float* __restrict__ y, float* bnsum, float* bnss, int nrows) {
    int d = threadIdx.x; // 0..127
    int row0 = blockIdx.x * 64;
    float bgd = bg[d];
    float s = 0.0f, ss = 0.0f;
    for (int i = 0; i < 64; ++i) {
        int nrow = row0 + i;
        if (nrow >= nrows) break;
        size_t idx = (size_t)nrow * DD + d;
        float c = fmaxf(agg[idx] * norm_dst[nrow] + bgd, 0.0f);
        float val = c + res[idx];
        y[idx] = val;
        s += val;
        ss += val * val;
    }
    atomicAdd(&bnsum[d], s);
    atomicAdd(&bnss[d], ss);
}

__global__ void k_bn_finalize(const float* bnsum, const float* bnss,
                              const float* __restrict__ gamma, const float* __restrict__ beta,
                              float* scale, float* shift, float invN) {
    int d = threadIdx.x;
    float mu = bnsum[d] * invN;
    float var = bnss[d] * invN - mu * mu;
    float rs = rsqrtf(var + BN_EPS);
    float sc = rs * gamma[d];
    scale[d] = sc;
    shift[d] = beta[d] - mu * sc;
}

__global__ void k_bn_apply(const float* __restrict__ y, const float* __restrict__ scale,
                           const float* __restrict__ shift, float* __restrict__ h, int total) {
    int idx = blockIdx.x * blockDim.x + threadIdx.x;
    if (idx >= total) return;
    int d = idx & (DD - 1);
    h[idx] = y[idx] * scale[d] + shift[d];
}

// ---------------- pooling ----------------
__global__ void k_pool_scatter(const float* __restrict__ h, const int* __restrict__ ids,
                               float* sum, float* cnt) {
    int n = blockIdx.x, d = threadIdx.x;
    int g = ids[n];
    atomicAdd(&sum[(size_t)g * DD + d], h[(size_t)n * DD + d]);
    if (d == 0) atomicAdd(&cnt[g], 1.0f);
}

__global__ void k_finalize_graph(const float* __restrict__ gsum, const float* __restrict__ gcnt,
                                 float* __restrict__ out_gf, __bf16* __restrict__ gfb) {
    int g = blockIdx.x, d = threadIdx.x;
    float c = fmaxf(gcnt[g], 1.0f);
    float v = gsum[(size_t)g * DD + d] / c;
    out_gf[(size_t)g * DD + d] = v;
    gfb[(size_t)g * DD + d] = (__bf16)v;
}

__global__ void k_finalize_motif(const float* __restrict__ msum, const float* __restrict__ mcnt,
                                 __bf16* __restrict__ hsubb) {
    int m = blockIdx.x, d = threadIdx.x; // m = 0..MM-1 -> segment m+1
    int row = m + 1;
    float c = fmaxf(mcnt[row], 1.0f);
    hsubb[(size_t)m * DD + d] = (__bf16)(msum[(size_t)row * DD + d] / c);
}

// ---------------- host launcher ----------------

extern "C" void kernel_launch(void* const* d_in, const int* in_sizes, int n_in,
                              void* d_out, int out_size, void* d_ws, size_t ws_size,
                              hipStream_t stream) {
    const float* node_feats = (const float*)d_in[0];
    const int*   src        = (const int*)d_in[1];
    const int*   dst        = (const int*)d_in[2];
    const int*   graph_ids  = (const int*)d_in[3];
    const int*   motif_b    = (const int*)d_in[4];
    const float* Wg   = (const float*)d_in[5];
    const float* bg   = (const float*)d_in[6];
    const float* Wr   = (const float*)d_in[7];
    const float* br   = (const float*)d_in[8];
    const float* gamma= (const float*)d_in[9];
    const float* beta = (const float*)d_in[10];
    const float* Wfeat= (const float*)d_in[11];
    const float* bfeat= (const float*)d_in[12];
    const float* W1   = (const float*)d_in[13];
    const float* b1   = (const float*)d_in[14];
    const float* W2   = (const float*)d_in[15];
    const float* b2   = (const float*)d_in[16];

    float* out = (float*)d_out;
    float* out_gf  = out;                 // [512,128]
    float* out_glb = out + GG * DD;       // [512,128]
    float* out_sub = out + 2 * GG * DD;   // [2048,128]

    // workspace carve-up
    size_t off = 0;
    char* base = (char*)d_ws;
    auto carve = [&](size_t bytes) -> char* {
        char* p = base + off;
        off += (bytes + 255) & ~(size_t)255;
        return p;
    };
    float* norm_src = (float*)carve(NN * 4);
    float* norm_dst = (float*)carve(NN * 4);
    float* h    = (float*)carve((size_t)NN * DD * 4);
    float* hw   = (float*)carve((size_t)NN * DD * 4);   // reused as y
    float* res  = (float*)carve((size_t)NN * DD * 4);
    float* agg  = (float*)carve((size_t)NN * DD * 4);
    __bf16* hb  = (__bf16*)carve((size_t)NN * DD * 2);
    __bf16* hsb = (__bf16*)carve((size_t)NN * DD * 2);
    __bf16* WgT = (__bf16*)carve((size_t)LL * DD * DD * 2);
    __bf16* WrT = (__bf16*)carve((size_t)LL * DD * DD * 2);
    __bf16* WfT = (__bf16*)carve((size_t)FFN * DD * 2);
    __bf16* W1T = (__bf16*)carve((size_t)FFN * FFN * 2);
    __bf16* W2T = (__bf16*)carve((size_t)(FFN / 2) * FFN * 2);
    float* bnsum   = (float*)carve(DD * 4);
    float* bnss    = (float*)carve(DD * 4);
    float* bnscale = (float*)carve(DD * 4);
    float* bnshift = (float*)carve(DD * 4);
    float* gsum = (float*)carve((size_t)GG * DD * 4);
    float* gcnt = (float*)carve(GG * 4);
    float* msum = (float*)carve((size_t)(MM + 1) * DD * 4);
    float* mcnt = (float*)carve((MM + 1) * 4);
    __bf16* gfb   = (__bf16*)carve((size_t)GG * DD * 2);
    __bf16* hsubb = (__bf16*)carve((size_t)MM * DD * 2);
    float*  fg  = (float*)carve((size_t)GG * FFN * 4);
    __bf16* fgb = (__bf16*)carve((size_t)GG * FFN * 2);
    float*  zg  = (float*)carve((size_t)GG * FFN * 4);
    __bf16* zgb = (__bf16*)carve((size_t)GG * FFN * 2);
    float*  fs  = (float*)carve((size_t)MM * FFN * 4);
    __bf16* fsb = (__bf16*)carve((size_t)MM * FFN * 2);
    float*  zs  = (float*)carve((size_t)MM * FFN * 4);
    __bf16* zsb = (__bf16*)carve((size_t)MM * FFN * 2);
    (void)ws_size; (void)in_sizes; (void)n_in; (void)out_size;

    const int T = 256;
    const size_t ND = (size_t)NN * DD;

    // --- degrees -> norms ---
    k_zero_f32<<<CDIV(2 * NN, T), T, 0, stream>>>(norm_src, (size_t)2 * NN); // norm_dst starts 48 floats after norm_src+NN (256-B pad)
    k_deg_accum<<<CDIV(EE, T), T, 0, stream>>>(src, dst, norm_src, norm_dst, EE);
    k_rsqrt_clip<<<CDIV(2 * NN, T), T, 0, stream>>>(norm_src, 2 * NN);

    // --- weight prep (fp32 -> bf16 transposed) ---
    for (int l = 0; l < LL; ++l) {
        k_transpose_bf16<<<CDIV(DD * DD, T), T, 0, stream>>>(Wg + (size_t)l * DD * DD, WgT + (size_t)l * DD * DD, DD, DD);
        k_transpose_bf16<<<CDIV(DD * DD, T), T, 0, stream>>>(Wr + (size_t)l * DD * DD, WrT + (size_t)l * DD * DD, DD, DD);
    }
    k_transpose_bf16<<<CDIV(DD * FFN, T), T, 0, stream>>>(Wfeat, WfT, DD, FFN);
    k_transpose_bf16<<<CDIV(FFN * FFN, T), T, 0, stream>>>(W1, W1T, FFN, FFN);
    k_transpose_bf16<<<CDIV(FFN * (FFN / 2), T), T, 0, stream>>>(W2, W2T, FFN, FFN / 2);

    // GEMM launch helper: waves = ceil(Mtiles/MT) * (Ncols/16), 4 waves/block
    auto blocks_for = [](int Mtiles, int MT, int Ncols) {
        int waves = CDIV(Mtiles, MT) * (Ncols >> 4);
        return CDIV(waves, 4);
    };

    // --- GCN layers ---
    const int Mtiles_N = NN / 16;                 // 3125
    int blk_big = blocks_for(Mtiles_N, 8, DD);    // 782 blocks, 3128 waves
    for (int l = 0; l < LL; ++l) {
        const float* hin = (l == 0) ? node_feats : h;
        k_prep_h<<<CDIV((int)ND, T), T, 0, stream>>>(hin, norm_src, hb, hsb, (int)ND);

        // res = relu(h @ Wr + br)
        k_wmma_gemm_t<4, 8, true, false><<<blk_big, 128, 0, stream>>>(
            hb, WrT + (size_t)l * DD * DD, br + (size_t)l * DD, res, nullptr, Mtiles_N, DD);
        // hw = (h*norm_src) @ Wg
        k_wmma_gemm_t<4, 8, false, false><<<blk_big, 128, 0, stream>>>(
            hsb, WgT + (size_t)l * DD * DD, nullptr, hw, nullptr, Mtiles_N, DD);

        // agg = scatter_sum(hw[src], dst)
        k_zero_f32<<<CDIV((int)ND, T), T, 0, stream>>>(agg, ND);
        k_edge_scatter<<<CDIV(EE * 32, T), T, 0, stream>>>(src, dst, hw, agg, EE);

        // y (into hw) + BN stats
        k_zero_f32<<<1, T, 0, stream>>>(bnsum, (size_t)2 * DD); // bnsum+bnss contiguous
        k_combine_bnstats<<<CDIV(NN, 64), DD, 0, stream>>>(agg, res, norm_dst,
                                                           bg + (size_t)l * DD,
                                                           hw, bnsum, bnss, NN);
        k_bn_finalize<<<1, DD, 0, stream>>>(bnsum, bnss, gamma + (size_t)l * DD,
                                            beta + (size_t)l * DD, bnscale, bnshift,
                                            1.0f / (float)NN);
        k_bn_apply<<<CDIV((int)ND, T), T, 0, stream>>>(hw, bnscale, bnshift, h, (int)ND);
    }

    // --- pooling ---
    k_zero_f32<<<CDIV(GG * DD + GG, T), T, 0, stream>>>(gsum, (size_t)GG * DD + GG);
    k_zero_f32<<<CDIV((MM + 1) * DD + (MM + 1), T), T, 0, stream>>>(msum, (size_t)(MM + 1) * DD + (MM + 1));
    k_pool_scatter<<<NN, DD, 0, stream>>>(h, graph_ids, gsum, gcnt);
    k_pool_scatter<<<NN, DD, 0, stream>>>(h, motif_b, msum, mcnt);
    k_finalize_graph<<<GG, DD, 0, stream>>>(gsum, gcnt, out_gf, gfb);
    k_finalize_motif<<<MM, DD, 0, stream>>>(msum, mcnt, hsubb);

    // --- head(graph_feats) -> out_global ---
    {
        const int MtG = GG / 16; // 32
        k_wmma_gemm_t<4, 2, false, true><<<blocks_for(MtG, 2, FFN), 128, 0, stream>>>(
            gfb, WfT, bfeat, fg, fgb, MtG, FFN);
        k_wmma_gemm_t<8, 2, true, true><<<blocks_for(MtG, 2, FFN), 128, 0, stream>>>(
            fgb, W1T, b1, zg, zgb, MtG, FFN);
        k_wmma_gemm_t<8, 2, false, false><<<blocks_for(MtG, 2, FFN / 2), 128, 0, stream>>>(
            zgb, W2T, b2, out_glb, nullptr, MtG, FFN / 2);
    }
    // --- head(h_sub) -> out_sub ---
    {
        const int MtS = MM / 16; // 128
        k_wmma_gemm_t<4, 4, false, true><<<blocks_for(MtS, 4, FFN), 128, 0, stream>>>(
            hsubb, WfT, bfeat, fs, fsb, MtS, FFN);
        k_wmma_gemm_t<8, 4, true, true><<<blocks_for(MtS, 4, FFN), 128, 0, stream>>>(
            fsb, W1T, b1, zs, zsb, MtS, FFN);
        k_wmma_gemm_t<8, 4, false, false><<<blocks_for(MtS, 4, FFN / 2), 128, 0, stream>>>(
            zsb, W2T, b2, out_sub, nullptr, MtS, FFN / 2);
    }
}
